// MultiHeadedAttention_73967926772096
// MI455X (gfx1250) — matrix operand order repaired and checked by reading, compiled, not run on hardware
//
#include <hip/hip_runtime.h>
#include <hip/hip_bf16.h>

// ---------------------------------------------------------------------------
// MHA forward for gfx1250 (MI455X): all GEMMs + attention via
// v_wmma_f32_16x16x32_f16 (f16 inputs, f32 accumulation).
// v5: launch_bounds + no unroll on key loop (kill scratch spills seen in v4),
//     step-major cross-lane reductions (8 ds_bpermute in flight per wait
//     instead of 64 serialized wait-0s), keep sched_barrier load/WMMA fences,
//     bit-packed mask, LDS tile staging, base-2 softmax.
// ---------------------------------------------------------------------------

typedef __attribute__((ext_vector_type(16))) _Float16 v16h;
typedef __attribute__((ext_vector_type(8)))  _Float16 v8h;
typedef __attribute__((ext_vector_type(8)))  float    v8f;
typedef __attribute__((ext_vector_type(4)))  int      v4i;

#define BB 2
#define SS 2048
#define DD 1024
#define HH 16
#define DKK 64
#define MM (BB * SS)          // 4096 tokens
#define SW (SS / 32)          // packed-mask words per row

#define LOG2E 1.44269504088896340736f

// sched_barrier mask: everything may cross EXCEPT DS (0x80/0x100/0x200) and
// MFMA/WMMA (0x8) -> loads stay above the fence, WMMAs stay below it.
#define FENCE_DS_WMMA() __builtin_amdgcn_sched_barrier(0x477)

static __device__ __forceinline__ v16h ld_frag16(const _Float16* p) {
    v16h r;
    *(v8h*)&r       = *(const v8h*)p;        // elements 0..7
    *((v8h*)&r + 1) = *(const v8h*)(p + 8);  // elements 8..15
    return r;
}

// ---------------------------------------------------------------------------
// Weight transpose + f32 -> f16 convert: Wt[n][k] = (f16) W[k][n]
// ---------------------------------------------------------------------------
__global__ void wt_transpose_cvt(const float* __restrict__ W, _Float16* __restrict__ Wt) {
    int idx = blockIdx.x * blockDim.x + threadIdx.x;    // over D*D
    if (idx >= DD * DD) return;
    int n = idx / DD;
    int k = idx - n * DD;
    Wt[(size_t)n * DD + k] = (_Float16)W[(size_t)k * DD + n];
}

// ---------------------------------------------------------------------------
// Pack int32 mask [B,S,S] into 1 bit/key: pmask[b*S+q][S/32] words.
// ---------------------------------------------------------------------------
__global__ void mask_pack(const int* __restrict__ mask, unsigned* __restrict__ pmask) {
    int w = blockIdx.x * blockDim.x + threadIdx.x;      // over B*S*(S/32)
    if (w >= BB * SS * SW) return;
    const v4i* mp = (const v4i*)(mask + (size_t)w * 32);
    unsigned bits = 0;
#pragma unroll
    for (int q = 0; q < 8; ++q) {
        v4i m = mp[q];
        bits |= (m[0] != 0 ? 1u : 0u) << (4 * q + 0);
        bits |= (m[1] != 0 ? 1u : 0u) << (4 * q + 1);
        bits |= (m[2] != 0 ? 1u : 0u) << (4 * q + 2);
        bits |= (m[3] != 0 ? 1u : 0u) << (4 * q + 3);
    }
    pmask[w] = bits;
}

// ---------------------------------------------------------------------------
// GEMM: C[M=4096][N=1024] = A[M][1024] * W + bias, Wt = W^T f16 [N][K].
// Block: 256 thr = 8 waves; each wave owns a 16(M)x64(N) strip; the 64x32
// B tile is shared by all 8 waves -> staged once per block in LDS.
// mode 0: f16 -> [B,H,S,DK]; mode 1: f16 -> [B,H,DK,S]; mode 2: f32 -> [B,S,D]
// ---------------------------------------------------------------------------
template <typename AT>
__global__ void __launch_bounds__(256, 1)
gemm_wmma(const AT* __restrict__ A, const _Float16* __restrict__ Wt,
          const float* __restrict__ bias, void* __restrict__ out,
          int mode) {
    __shared__ __align__(16) _Float16 bs[64 * 32];   // 4KB B tile, shared by 8 waves

    const int lane = threadIdx.x & 31;
    const int wave = threadIdx.x >> 5;
    const int lm   = lane & 15;
    const int half = lane >> 4;

    const int m0 = blockIdx.y * 128 + wave * 16;   // 8 waves * 16 rows
    const int n0 = blockIdx.x * 64;

    v8f acc[4] = {};

    const int arow = m0 + lm;
    const AT* arowp = A + (size_t)arow * DD;

    // cooperative-staging coordinates: 256 threads x 16B = 4KB tile
    const int sn = threadIdx.x >> 2;          // 0..63 (n within tile)
    const int sc = (threadIdx.x & 3) * 8;     // 0..24 (k within tile, 8-half runs)
    const _Float16* wrow = Wt + (size_t)(n0 + sn) * DD + sc;

    for (int k = 0; k < DD; k += 32) {
        __syncthreads();
        *(v8h*)&bs[sn * 32 + sc] = *(const v8h*)(wrow + k);
        if (k + 32 < DD) __builtin_prefetch(wrow + k + 32, 0, 3);
        __syncthreads();

        // ---- A fragment (16x32, contiguous k = half*16 + e, same order as B) ----
        v16h a;
        const AT* ap = arowp + k + half * 16;
#pragma unroll
        for (int e = 0; e < 8; ++e) {
            a[e]     = (_Float16)ap[e];        // K = k + half*16 + e
            a[8 + e] = (_Float16)ap[8 + e];    // K = k + half*16 + 8 + e
        }
        if (k + 32 < DD) __builtin_prefetch(arowp + k + 32, 0, 3);

        // ---- all 4 B fragments, fence, then 4 independent WMMAs ----
        v16h bf0 = ld_frag16(&bs[(0 * 16 + lm) * 32 + half * 16]);
        v16h bf1 = ld_frag16(&bs[(1 * 16 + lm) * 32 + half * 16]);
        v16h bf2 = ld_frag16(&bs[(2 * 16 + lm) * 32 + half * 16]);
        v16h bf3 = ld_frag16(&bs[(3 * 16 + lm) * 32 + half * 16]);
        FENCE_DS_WMMA();
        acc[0] = __builtin_amdgcn_wmma_f32_16x16x32_f16(false, a, false, bf0, (short)0, acc[0], false, false);
        acc[1] = __builtin_amdgcn_wmma_f32_16x16x32_f16(false, a, false, bf1, (short)0, acc[1], false, false);
        acc[2] = __builtin_amdgcn_wmma_f32_16x16x32_f16(false, a, false, bf2, (short)0, acc[2], false, false);
        acc[3] = __builtin_amdgcn_wmma_f32_16x16x32_f16(false, a, false, bf3, (short)0, acc[3], false, false);
    }

    // ---- epilogue: bias + scatter per layout ----
#pragma unroll
    for (int nt = 0; nt < 4; ++nt) {
        const int nn = n0 + nt * 16 + lm;        // output channel
        const float bv = bias[nn];
        const int hh = nn >> 6;                  // head
        const int dd = nn & 63;                  // within-head dim
#pragma unroll
        for (int i = 0; i < 8; ++i) {
            const int mrow = m0 + i + half * 8;  // token
            const int b = mrow >> 11;            // / S
            const int s = mrow & (SS - 1);
            const float val = acc[nt][i] + bv;
            if (mode == 0) {
                ((_Float16*)out)[(((size_t)b * HH + hh) * SS + s) * DKK + dd] = (_Float16)val;
            } else if (mode == 1) {
                ((_Float16*)out)[(((size_t)b * HH + hh) * DKK + dd) * SS + s] = (_Float16)val;
            } else {
                ((float*)out)[(size_t)mrow * DD + nn] = val;
            }
        }
    }
}

// ---------------------------------------------------------------------------
// Fused flash attention: one wave per 16-query tile per (b,h); 8 waves/block
// share (b,h) so K/V chunks are staged once per block in LDS.
// Qh,Kh: [B,H,S,DK] f16.  Vt: [B,H,DK,S] f16.  attn out: [B,S,D] f16.
// Key loop step 32 (two 16x16 score tiles -> one 16x32 P A-fragment).
// Softmax in base 2: scores pre-scaled by log2(e)/sqrt(DK).
// ---------------------------------------------------------------------------
__global__ void __launch_bounds__(256, 1)
flash_attn_wmma(const _Float16* __restrict__ Qh,
                const _Float16* __restrict__ Kh,
                const _Float16* __restrict__ Vt,
                const unsigned* __restrict__ pmask,
                _Float16* __restrict__ attn) {
    __shared__ __align__(16) _Float16 Ks[32 * 64];    // 4KB K chunk (32 keys x 64 d)
    __shared__ __align__(16) _Float16 Vs[64 * 32];    // 4KB V chunk (64 d x 32 keys)
    __shared__ __align__(16) _Float16 pls[8][16 * 32];// per-wave P staging (1KB/wave)

    const int lane = threadIdx.x & 31;
    const int wave = threadIdx.x >> 5;
    const int lm   = lane & 15;
    const int half = lane >> 4;

    const int b  = blockIdx.z;
    const int h  = blockIdx.y;
    const int q0 = (blockIdx.x * 8 + wave) * 16;

    const _Float16* Qbase  = Qh + ((size_t)b * HH + h) * SS * DKK;
    const _Float16* Kbase  = Kh + ((size_t)b * HH + h) * SS * DKK;
    const _Float16* Vtbase = Vt + ((size_t)b * HH + h) * DKK * SS;

    // ---- Q A-fragments for d=0..31 and d=32..63 ----
    v16h aq0, aq1;
    {
        const _Float16* qrow = Qbase + (size_t)(q0 + lm) * DKK;
        const _Float16* q0p = qrow + half * 16;
        const _Float16* q1p = qrow + 32 + half * 16;
#pragma unroll
        for (int e = 0; e < 8; ++e) {
            aq0[e] = q0p[e];  aq0[8 + e] = q0p[8 + e];
            aq1[e] = q1p[e];  aq1[8 + e] = q1p[8 + e];
        }
    }

    float rm[8], rl[8];
#pragma unroll
    for (int i = 0; i < 8; ++i) { rm[i] = -3.0e38f; rl[i] = 0.0f; }
    v8f acc[4] = {};

    _Float16* pl = &pls[wave][0];

    // cooperative-staging coordinates (256 threads x 16B = 4KB per tile)
    const int kr = threadIdx.x >> 3;          // 0..31 key row
    const int kc = (threadIdx.x & 7) * 8;     // 0..56 d col
    const int vr = threadIdx.x >> 2;          // 0..63 d row
    const int vc = (threadIdx.x & 3) * 8;     // 0..24 key col
    const _Float16* ksrc = Kbase + (size_t)kr * DKK + kc;
    const _Float16* vsrc = Vtbase + (size_t)vr * SS + vc;

    const float cs = 0.125f * LOG2E;          // scale * log2(e)
    const unsigned* mrowp = pmask + (size_t)(b * SS + q0 + half * 8) * SW;

#pragma clang loop unroll(disable)
    for (int j0 = 0; j0 < SS; j0 += 32) {
        __syncthreads();
        *(v8h*)&Ks[kr * 64 + kc] = *(const v8h*)(ksrc + (size_t)j0 * DKK);
        *(v8h*)&Vs[vr * 32 + vc] = *(const v8h*)(vsrc + j0);
        if (j0 + 32 < SS) {
            __builtin_prefetch(ksrc + (size_t)(j0 + 32) * DKK, 0, 3);
            __builtin_prefetch(vsrc + j0 + 32, 0, 3);
        }
        __syncthreads();

        // ---- all 4 K fragments, fence, 4 WMMAs on 2 independent chains ----
        const _Float16* kr0 = &Ks[(0 * 16 + lm) * 64];
        const _Float16* kr1 = &Ks[(1 * 16 + lm) * 64];
        v16h bk00 = ld_frag16(kr0 + half * 16);        // nt0, d =  0..31
        v16h bk01 = ld_frag16(kr0 + 32 + half * 16);   // nt0, d = 32..63
        v16h bk10 = ld_frag16(kr1 + half * 16);        // nt1, d =  0..31
        v16h bk11 = ld_frag16(kr1 + 32 + half * 16);   // nt1, d = 32..63
        FENCE_DS_WMMA();
        v8f z0 = {}, z1 = {};
        z0 = __builtin_amdgcn_wmma_f32_16x16x32_f16(false, aq0, false, bk00, (short)0, z0, false, false);
        z1 = __builtin_amdgcn_wmma_f32_16x16x32_f16(false, aq0, false, bk10, (short)0, z1, false, false);
        z0 = __builtin_amdgcn_wmma_f32_16x16x32_f16(false, aq1, false, bk01, (short)0, z0, false, false);
        z1 = __builtin_amdgcn_wmma_f32_16x16x32_f16(false, aq1, false, bk11, (short)0, z1, false, false);

        // ---- scale + bit-mask ----
        float s0v[8], s1v[8], t[8];
#pragma unroll
        for (int i = 0; i < 8; ++i) {
            const unsigned mw = mrowp[(size_t)i * SW + (j0 >> 5)];
            float s0 = z0[i] * cs;
            float s1 = z1[i] * cs;
            if (!((mw >> lm) & 1u))        s0 = -1.443e9f;   // -1e9 * log2(e)
            if (!((mw >> (16 + lm)) & 1u)) s1 = -1.443e9f;
            s0v[i] = s0; s1v[i] = s1;
            t[i] = fmaxf(s0, s1);
        }

        // ---- row max: step-major butterfly (8 independent DS ops per wait) ----
#pragma unroll
        for (int step = 1; step <= 8; step <<= 1) {
            float u[8];
#pragma unroll
            for (int i = 0; i < 8; ++i) u[i] = __shfl_xor(t[i], step, 32);
#pragma unroll
            for (int i = 0; i < 8; ++i) t[i] = fmaxf(t[i], u[i]);
        }

        // ---- rescale state, exponentials, stage P to per-wave LDS ----
        float rs[8], cf[8];
#pragma unroll
        for (int i = 0; i < 8; ++i) {
            const float mnew = fmaxf(rm[i], t[i]);
            cf[i] = exp2f(rm[i] - mnew);
            rm[i] = mnew;
            const float p0 = exp2f(s0v[i] - mnew);
            const float p1 = exp2f(s1v[i] - mnew);
            rs[i] = p0 + p1;
            const int row = i + half * 8;
            pl[row * 32 + lm]      = (_Float16)p0;
            pl[row * 32 + 16 + lm] = (_Float16)p1;
            acc[0][i] *= cf[i]; acc[1][i] *= cf[i];
            acc[2][i] *= cf[i]; acc[3][i] *= cf[i];
        }

        // ---- row sum: step-major butterfly ----
#pragma unroll
        for (int step = 1; step <= 8; step <<= 1) {
            float u[8];
#pragma unroll
            for (int i = 0; i < 8; ++i) u[i] = __shfl_xor(rs[i], step, 32);
#pragma unroll
            for (int i = 0; i < 8; ++i) rs[i] += u[i];
        }
#pragma unroll
        for (int i = 0; i < 8; ++i) rl[i] = rl[i] * cf[i] + rs[i];

        // ---- P A-fragment from per-wave LDS (same-wave DS ops are in-order) ----
        v16h pa;
        {
            const _Float16* pp = pl + lm * 32 + half * 16;
            *(v8h*)&pa       = *(const v8h*)pp;
            *((v8h*)&pa + 1) = *(const v8h*)(pp + 8);
        }

        // ---- all 4 V fragments, fence, 4 independent WMMA chains ----
        v16h bv0 = ld_frag16(&Vs[(0 * 16 + lm) * 32 + half * 16]);
        v16h bv1 = ld_frag16(&Vs[(1 * 16 + lm) * 32 + half * 16]);
        v16h bv2 = ld_frag16(&Vs[(2 * 16 + lm) * 32 + half * 16]);
        v16h bv3 = ld_frag16(&Vs[(3 * 16 + lm) * 32 + half * 16]);
        FENCE_DS_WMMA();
        acc[0] = __builtin_amdgcn_wmma_f32_16x16x32_f16(false, pa, false, bv0, (short)0, acc[0], false, false);
        acc[1] = __builtin_amdgcn_wmma_f32_16x16x32_f16(false, pa, false, bv1, (short)0, acc[1], false, false);
        acc[2] = __builtin_amdgcn_wmma_f32_16x16x32_f16(false, pa, false, bv2, (short)0, acc[2], false, false);
        acc[3] = __builtin_amdgcn_wmma_f32_16x16x32_f16(false, pa, false, bv3, (short)0, acc[3], false, false);
    }

    // ---- normalize + store merged-head layout [B,S,D] f16 ----
#pragma unroll
    for (int dt = 0; dt < 4; ++dt) {
        const int c = h * DKK + dt * 16 + lm;
#pragma unroll
        for (int i = 0; i < 8; ++i) {
            const int mrow = q0 + i + half * 8;
            const float o = acc[dt][i] / rl[i];
            attn[((size_t)b * SS + mrow) * DD + c] = (_Float16)o;
        }
    }
}

// ---------------------------------------------------------------------------
// Host-side launch
// ---------------------------------------------------------------------------
extern "C" void kernel_launch(void* const* d_in, const int* in_sizes, int n_in,
                              void* d_out, int out_size, void* d_ws, size_t ws_size,
                              hipStream_t stream) {
    const float* query = (const float*)d_in[0];
    const float* key_  = (const float*)d_in[1];
    const float* value = (const float*)d_in[2];
    const int*   mask  = (const int*)  d_in[3];
    const float* Wq = (const float*)d_in[4];  const float* bq = (const float*)d_in[5];
    const float* Wk = (const float*)d_in[6];  const float* bk = (const float*)d_in[7];
    const float* Wv = (const float*)d_in[8];  const float* bv = (const float*)d_in[9];
    const float* Wo = (const float*)d_in[10]; const float* bo = (const float*)d_in[11];
    float* out = (float*)d_out;

    // workspace layout, total 41 MiB
    char* ws = (char*)d_ws;
    size_t off = 0;
    _Float16* Wq_t = (_Float16*)(ws + off); off += (size_t)DD * DD * 2;
    _Float16* Wk_t = (_Float16*)(ws + off); off += (size_t)DD * DD * 2;
    _Float16* Wv_t = (_Float16*)(ws + off); off += (size_t)DD * DD * 2;
    _Float16* Wo_t = (_Float16*)(ws + off); off += (size_t)DD * DD * 2;
    _Float16* Qh   = (_Float16*)(ws + off); off += (size_t)MM * DD * 2;
    _Float16* Kh   = (_Float16*)(ws + off); off += (size_t)MM * DD * 2;
    _Float16* Vt   = (_Float16*)(ws + off); off += (size_t)MM * DD * 2;
    _Float16* attn = (_Float16*)(ws + off); off += (size_t)MM * DD * 2;
    unsigned* pmask = (unsigned*)(ws + off); off += (size_t)BB * SS * SW * 4;

    // 1) transpose/convert weights + pack mask
    const int tgrid = (DD * DD + 255) / 256;
    wt_transpose_cvt<<<tgrid, 256, 0, stream>>>(Wq, Wq_t);
    wt_transpose_cvt<<<tgrid, 256, 0, stream>>>(Wk, Wk_t);
    wt_transpose_cvt<<<tgrid, 256, 0, stream>>>(Wv, Wv_t);
    wt_transpose_cvt<<<tgrid, 256, 0, stream>>>(Wo, Wo_t);
    mask_pack<<<(BB * SS * SW + 255) / 256, 256, 0, stream>>>(mask, pmask);

    // 2) Q/K/V projections (A = f32 inputs, converted in-register)
    dim3 ggrid(DD / 64, MM / 128);
    gemm_wmma<float><<<ggrid, 256, 0, stream>>>(query, Wq_t, bq, (void*)Qh, 0);
    gemm_wmma<float><<<ggrid, 256, 0, stream>>>(key_,  Wk_t, bk, (void*)Kh, 0);
    gemm_wmma<float><<<ggrid, 256, 0, stream>>>(value, Wv_t, bv, (void*)Vt, 1);

    // 3) fused flash attention: grid (S/16/8, H, B)
    dim3 agrid(SS / 16 / 8, HH, BB);
    flash_attn_wmma<<<agrid, 256, 0, stream>>>(Qh, Kh, Vt, pmask, attn);

    // 4) output projection -> f32 d_out
    gemm_wmma<_Float16><<<ggrid, 256, 0, stream>>>(attn, Wo_t, bo, (void*)out, 2);
}
